// Predictor_6545530159156
// MI455X (gfx1250) — matrix operand order repaired and checked by reading, not compiled, not run on hardware
//
#include <hip/hip_runtime.h>
#include <hip/hip_bf16.h>

// ---------------- types ----------------
typedef _Float16 h16;
typedef __attribute__((ext_vector_type(16))) _Float16 v16h;
typedef __attribute__((ext_vector_type(2)))  __fp16   f16x2;   // cvt_pkrtz return type
typedef __attribute__((ext_vector_type(8)))  float    v8f;

#define NUM_LOC_  20000
#define NUM_USER_ 5000
#define NUM_PROTO_ 64
#define DD_   128
#define BB_   64
#define LL_   128
#define DM_   256
#define NH_   2
#define DH_   128
#define E_LL_ 400000
#define E_UL_ 200000
#define N_BI_ (NUM_LOC_ + NUM_USER_ + NUM_PROTO_)   // 25064
#define N_PL_ (NUM_LOC_ + NUM_PROTO_)               // 20064

// GEMM flags
#define GF_RELU   1
#define GF_ACCUM  2
#define GF_RELU_A 4

struct GemmDesc {
    const float* A; const float* B; float* C; const float* bias;
    long sam, sak, sbk, sbn, scm, scn;          // element strides
    long bAy, bAz, bBy, bBz, bCy, bCz;          // batch strides (blockIdx.y / .z)
    int M, N, K, flags;
    float alpha;
};

union V16U { v16h v; f16x2 p[8]; };

// ---------------------------------------------------------------------------
// One 16x64 strip of C per wave: 1 A-fragment feeds 4 v_wmma issues per K-step.
// Fragments gathered directly from global. AK1/BK1: K-contiguous operands use
// float4 (global_load_b128) loads with immediate offsets. MASKED variant uses
// clamped addresses + branchless cndmask zeroing (no exec-mask branches).
// K is assumed to be a multiple of 32 (true for every GEMM in this model).
// ---------------------------------------------------------------------------
template<int FLAGS, bool MASKED, bool AK1, bool BK1>
__device__ __forceinline__ void gemm_strip(const GemmDesc& d,
    const float* A, const float* B, float* C, int m0, int n0, int lane)
{
    const int r   = lane & 15;
    const int kb  = (lane >> 4) << 3;    // A: 0 or 8
    const int cc  = lane & 15;
    const int kbb = (lane >> 4) << 3;    // B: 0 or 8

    // hoisted A row pointer (+validity)
    int gm = m0 + r;
    bool mok = true;
    if (MASKED) { mok = gm < d.M; if (!mok) gm = d.M - 1; }
    const float* Arow = A + (long)gm * d.sam;

    // hoisted B column pointers (+validity)
    const float* Bcol[4];
    bool nok[4];
    #pragma unroll
    for (int t = 0; t < 4; ++t) {
        int col = n0 + (t << 4) + cc;
        nok[t] = true;
        if (MASKED) { nok[t] = col < d.N; if (!nok[t]) col = d.N - 1; }
        Bcol[t] = B + (long)col * d.sbn;
    }

    v8f acc0 = {}, acc1 = {}, acc2 = {}, acc3 = {};

    for (int k0 = 0; k0 < d.K; k0 += 32) {
        // ---- A fragment: halves 0..7 = K kb..kb+7, 8..15 = K kb+16..kb+23
        float av[16];
        if (AK1) {
            const float4 a0 = *(const float4*)(Arow + k0 + kb);
            const float4 a1 = *(const float4*)(Arow + k0 + kb + 4);
            const float4 a2 = *(const float4*)(Arow + k0 + kb + 16);
            const float4 a3 = *(const float4*)(Arow + k0 + kb + 20);
            av[0]=a0.x;  av[1]=a0.y;  av[2]=a0.z;  av[3]=a0.w;
            av[4]=a1.x;  av[5]=a1.y;  av[6]=a1.z;  av[7]=a1.w;
            av[8]=a2.x;  av[9]=a2.y;  av[10]=a2.z; av[11]=a2.w;
            av[12]=a3.x; av[13]=a3.y; av[14]=a3.z; av[15]=a3.w;
        } else {
            #pragma unroll
            for (int j = 0; j < 8; ++j) av[j]     = Arow[(long)(k0 + kb + j) * d.sak];
            #pragma unroll
            for (int j = 0; j < 8; ++j) av[8 + j] = Arow[(long)(k0 + kb + 16 + j) * d.sak];
        }
        #pragma unroll
        for (int j = 0; j < 16; ++j) {
            float v = av[j];
            if (FLAGS & GF_RELU_A) v = v > 0.0f ? v : 0.0f;
            if (MASKED) v = mok ? v : 0.0f;
            av[j] = v;
        }
        V16U ua;
        #pragma unroll
        for (int j = 0; j < 8; ++j)
            ua.p[j] = __builtin_amdgcn_cvt_pkrtz(av[2 * j], av[2 * j + 1]);

        // ---- 4 B fragments, 4 WMMAs
        #pragma unroll
        for (int t = 0; t < 4; ++t) {
            float bv[16];
            if (BK1) {
                const float4 b0 = *(const float4*)(Bcol[t] + k0 + kbb);
                const float4 b1 = *(const float4*)(Bcol[t] + k0 + kbb + 4);
                const float4 b2 = *(const float4*)(Bcol[t] + k0 + kbb + 16);
                const float4 b3 = *(const float4*)(Bcol[t] + k0 + kbb + 20);
                bv[0]=b0.x;  bv[1]=b0.y;  bv[2]=b0.z;  bv[3]=b0.w;
                bv[4]=b1.x;  bv[5]=b1.y;  bv[6]=b1.z;  bv[7]=b1.w;
                bv[8]=b2.x;  bv[9]=b2.y;  bv[10]=b2.z; bv[11]=b2.w;
                bv[12]=b3.x; bv[13]=b3.y; bv[14]=b3.z; bv[15]=b3.w;
            } else {
                #pragma unroll
                for (int j = 0; j < 16; ++j)
                    bv[j] = Bcol[t][(long)(k0 + kbb + j + (j & 8)) * d.sbk];
            }
            if (MASKED) {
                #pragma unroll
                for (int j = 0; j < 16; ++j) bv[j] = nok[t] ? bv[j] : 0.0f;
            }
            V16U ub;
            #pragma unroll
            for (int j = 0; j < 8; ++j)
                ub.p[j] = __builtin_amdgcn_cvt_pkrtz(bv[2 * j], bv[2 * j + 1]);

            if (t == 0) acc0 = __builtin_amdgcn_wmma_f32_16x16x32_f16(
                            false, ua.v, false, ub.v, (short)0, acc0, false, false);
            if (t == 1) acc1 = __builtin_amdgcn_wmma_f32_16x16x32_f16(
                            false, ua.v, false, ub.v, (short)0, acc1, false, false);
            if (t == 2) acc2 = __builtin_amdgcn_wmma_f32_16x16x32_f16(
                            false, ua.v, false, ub.v, (short)0, acc2, false, false);
            if (t == 3) acc3 = __builtin_amdgcn_wmma_f32_16x16x32_f16(
                            false, ua.v, false, ub.v, (short)0, acc3, false, false);
        }
    }

    // ---- epilogue: VGPR i -> M = i + 8*(lane>=16), N = lane&15
    const int rb = (lane >> 4) << 3;
    #pragma unroll
    for (int t = 0; t < 4; ++t) {
        const v8f& acc = t == 0 ? acc0 : t == 1 ? acc1 : t == 2 ? acc2 : acc3;
        const int n = n0 + (t << 4) + (lane & 15);
        if (MASKED && n >= d.N) continue;
        #pragma unroll
        for (int i = 0; i < 8; ++i) {
            const int m = m0 + rb + i;
            if (MASKED && m >= d.M) continue;
            float v = acc[i] * d.alpha;
            if (d.bias) v += d.bias[n];
            const long off = (long)m * d.scm + (long)n * d.scn;
            if (FLAGS & GF_ACCUM) v += C[off];
            if (FLAGS & GF_RELU)  v = v > 0.0f ? v : 0.0f;
            C[off] = v;
        }
    }
}

template<int FLAGS, bool AK1, bool BK1>
__launch_bounds__(256)
__global__ void gemm_wmma_kernel(GemmDesc d) {
    const int lane = threadIdx.x & 31;
    const int wib  = threadIdx.x >> 5;
    const int tilesM  = (d.M + 15) >> 4;
    const int tilesN  = (d.N + 15) >> 4;
    const int stripsN = (tilesN + 3) >> 2;
    const long total  = (long)tilesM * stripsN;
    const long waveId = (long)blockIdx.x * 8 + wib;
    const long stride = (long)gridDim.x * 8;

    const float* A = d.A + (long)blockIdx.y * d.bAy + (long)blockIdx.z * d.bAz;
    const float* B = d.B + (long)blockIdx.y * d.bBy + (long)blockIdx.z * d.bBz;
    float*       C = d.C + (long)blockIdx.y * d.bCy + (long)blockIdx.z * d.bCz;

    for (long s = waveId; s < total; s += stride) {
        const int tm = (int)(s / stripsN);
        const int sn = (int)(s % stripsN);
        const int m0 = tm << 4, n0 = sn << 6;
        if (m0 + 16 <= d.M && n0 + 64 <= d.N)
            gemm_strip<FLAGS, false, AK1, BK1>(d, A, B, C, m0, n0, lane);
        else
            gemm_strip<FLAGS, true, AK1, BK1>(d, A, B, C, m0, n0, lane);
    }
}

// ---------------------------------------------------------------------------
// Elementwise / reduction helper kernels (wave32)
// ---------------------------------------------------------------------------
__global__ void fill_kernel(float* p, long n, float v) {
    long g = (long)blockIdx.x * blockDim.x + threadIdx.x;
    for (; g < n; g += (long)gridDim.x * blockDim.x) p[g] = v;
}

// h = concat(loc_emb[1:], user_nr_emb, proto_emb)
__global__ void build_h_kernel(const float* loc, const float* unr,
                               const float* proto, float* h) {
    long g = (long)blockIdx.x * blockDim.x + threadIdx.x;
    const long total = (long)N_BI_ * DD_;
    for (; g < total; g += (long)gridDim.x * blockDim.x) {
        const int row = (int)(g >> 7), dd = (int)(g & 127);
        float v;
        if (row < NUM_LOC_)            v = loc[(long)(row + 1) * DD_ + dd];
        else if (row < NUM_LOC_ + NUM_USER_)
                                       v = unr[(long)(row - NUM_LOC_) * DD_ + dd];
        else                           v = proto[(long)(row - NUM_LOC_ - NUM_USER_) * DD_ + dd];
        h[g] = v;
    }
}

// agg[dst] += w * H[src + rowOff]   (float4 chunks, global f32 atomics)
__global__ void scatter_add_kernel(const int* src, const int* dst, const float* w,
                                   int E, const float* H, int rowOff, float* agg) {
    long g = (long)blockIdx.x * blockDim.x + threadIdx.x;
    const long total = (long)E * 32;
    for (; g < total; g += (long)gridDim.x * blockDim.x) {
        const int e = (int)(g >> 5);
        const int c = (int)(g & 31);
        const int sr = src[e] + rowOff;
        const int ds = dst[e];
        const float ww = w[e];
        const float4 hv = *(const float4*)(H + (long)sr * DD_ + c * 4);
        float* out = agg + (long)ds * DD_ + c * 4;
        atomicAdd(out + 0, ww * hv.x);
        atomicAdd(out + 1, ww * hv.y);
        atomicAdd(out + 2, ww * hv.z);
        atomicAdd(out + 3, ww * hv.w);
    }
}

// proto->user edges: unique dst per user, no atomics needed
__global__ void scatter_pu_kernel(const int* hidx, const float* puw,
                                  const float* h, float* agg) {
    const int u = blockIdx.x, dd = threadIdx.x;
    const float v = puw[u] * h[(long)(NUM_LOC_ + NUM_USER_ + hidx[u]) * DD_ + dd];
    agg[(long)(NUM_LOC_ + u) * DD_ + dd] += v;
}

// rows of 64 logits: argmax + pu_w = softmax peak = 1/sum(exp(l - max))
__global__ void proto_attn_kernel(const float* logits, float* puw, int* hidx) {
    const int row = blockIdx.x * (blockDim.x >> 5) + (threadIdx.x >> 5);
    if (row >= NUM_USER_) return;
    const int lane = threadIdx.x & 31;
    const float* Lr = logits + (long)row * 64;
    const float v0 = Lr[lane], v1 = Lr[lane + 32];
    float m = v0 >= v1 ? v0 : v1;
    int mi = v0 >= v1 ? lane : lane + 32;
    for (int o = 16; o > 0; o >>= 1) {
        const float om = __shfl_down(m, o);
        const int oi = __shfl_down(mi, o);
        if (om > m) { m = om; mi = oi; }
    }
    m = __shfl(m, 0); mi = __shfl(mi, 0);
    float s = __expf(v0 - m) + __expf(v1 - m);
    for (int o = 16; o > 0; o >>= 1) s += __shfl_down(s, o);
    if (lane == 0) { puw[row] = 1.0f / s; hidx[row] = mi; }
}

// out = in / max(||in||, 1e-12), D=128, one wave per row
__global__ void rownorm_kernel(const float* in, float* out, int rows) {
    const int row = blockIdx.x * (blockDim.x >> 5) + (threadIdx.x >> 5);
    if (row >= rows) return;
    const int lane = threadIdx.x & 31;
    const float4 a = ((const float4*)(in + (long)row * DD_))[lane];
    float ss = a.x * a.x + a.y * a.y + a.z * a.z + a.w * a.w;
    for (int o = 16; o > 0; o >>= 1) ss += __shfl_down(ss, o);
    ss = __shfl(ss, 0);
    const float inv = 1.0f / fmaxf(sqrtf(ss), 1e-12f);
    float4 r; r.x = a.x * inv; r.y = a.y * inv; r.z = a.z * inv; r.w = a.w * inv;
    ((float4*)(out + (long)row * DD_))[lane] = r;
}

// x[b,l,:] = concat(loc_emb[loc], time_emb[time]) * sqrt(DM) + pe[l]
__global__ void embed_kernel(const int* loc_seq, const int* time_seq,
                             const float* loc_emb, const float* time_emb,
                             const float* pe, float* x) {
    long g = (long)blockIdx.x * blockDim.x + threadIdx.x;
    const long total = (long)BB_ * LL_ * DM_;
    if (g >= total) return;
    const int dd = (int)(g & 255);
    const long bl = g >> 8;
    const int l = (int)(bl & 127);
    float e;
    if (dd < 128) e = loc_emb[(long)loc_seq[bl] * DD_ + dd];
    else          e = time_emb[(long)time_seq[bl] * DD_ + (dd - 128)];
    x[g] = e * 16.0f + pe[l * DM_ + dd];
}

// masked softmax over rows of s (rows = B*NH*L, cols = L = 128)
__global__ void attn_softmax_kernel(float* s, const int* valid_len) {
    const int row = blockIdx.x * (blockDim.x >> 5) + (threadIdx.x >> 5);
    if (row >= BB_ * NH_ * LL_) return;
    const int lane = threadIdx.x & 31;
    const int b = row >> 8;               // NH*L = 256 rows per batch
    const int vl = valid_len[b];
    float* p = s + (long)row * LL_;
    float4 v = ((float4*)p)[lane];
    const int k0 = lane << 2;
    v.x = (k0 + 0 < vl) ? v.x : -1e9f;
    v.y = (k0 + 1 < vl) ? v.y : -1e9f;
    v.z = (k0 + 2 < vl) ? v.z : -1e9f;
    v.w = (k0 + 3 < vl) ? v.w : -1e9f;
    float m = fmaxf(fmaxf(v.x, v.y), fmaxf(v.z, v.w));
    for (int o = 16; o > 0; o >>= 1) m = fmaxf(m, __shfl_down(m, o));
    m = __shfl(m, 0);
    float e0 = __expf(v.x - m), e1 = __expf(v.y - m),
          e2 = __expf(v.z - m), e3 = __expf(v.w - m);
    float su = e0 + e1 + e2 + e3;
    for (int o = 16; o > 0; o >>= 1) su += __shfl_down(su, o);
    su = __shfl(su, 0);
    const float inv = 1.0f / su;
    float4 r; r.x = e0 * inv; r.y = e1 * inv; r.z = e2 * inv; r.w = e3 * inv;
    ((float4*)p)[lane] = r;
}

// out = LN(x (+res)) * g + b   (DM=256, one wave per row)
__global__ void layernorm_kernel(const float* x, const float* res, float* out,
                                 const float* g, const float* bt, int rows) {
    const int row = blockIdx.x * (blockDim.x >> 5) + (threadIdx.x >> 5);
    if (row >= rows) return;
    const int lane = threadIdx.x & 31;
    const float4* xr = (const float4*)(x + (long)row * DM_);
    float4 a = xr[lane];
    float4 b = xr[lane + 32];
    if (res) {
        const float4* rr = (const float4*)(res + (long)row * DM_);
        const float4 r0 = rr[lane], r1 = rr[lane + 32];
        a.x += r0.x; a.y += r0.y; a.z += r0.z; a.w += r0.w;
        b.x += r1.x; b.y += r1.y; b.z += r1.z; b.w += r1.w;
    }
    float s = a.x + a.y + a.z + a.w + b.x + b.y + b.z + b.w;
    for (int o = 16; o > 0; o >>= 1) s += __shfl_down(s, o);
    const float mu = __shfl(s, 0) * (1.0f / 256.0f);
    float vs = (a.x - mu) * (a.x - mu) + (a.y - mu) * (a.y - mu) +
               (a.z - mu) * (a.z - mu) + (a.w - mu) * (a.w - mu) +
               (b.x - mu) * (b.x - mu) + (b.y - mu) * (b.y - mu) +
               (b.z - mu) * (b.z - mu) + (b.w - mu) * (b.w - mu);
    for (int o = 16; o > 0; o >>= 1) vs += __shfl_down(vs, o);
    const float var = __shfl(vs, 0) * (1.0f / 256.0f);
    const float inv = rsqrtf(var + 1e-5f);
    const int d0 = lane << 2, d1 = 128 + (lane << 2);
    float4 o0, o1;
    o0.x = (a.x - mu) * inv * g[d0 + 0] + bt[d0 + 0];
    o0.y = (a.y - mu) * inv * g[d0 + 1] + bt[d0 + 1];
    o0.z = (a.z - mu) * inv * g[d0 + 2] + bt[d0 + 2];
    o0.w = (a.w - mu) * inv * g[d0 + 3] + bt[d0 + 3];
    o1.x = (b.x - mu) * inv * g[d1 + 0] + bt[d1 + 0];
    o1.y = (b.y - mu) * inv * g[d1 + 1] + bt[d1 + 1];
    o1.z = (b.z - mu) * inv * g[d1 + 2] + bt[d1 + 2];
    o1.w = (b.w - mu) * inv * g[d1 + 3] + bt[d1 + 3];
    float4* orow = (float4*)(out + (long)row * DM_);
    orow[lane] = o0;
    orow[lane + 32] = o1;
}

// ctx[b] = concat(user_emb[uid-1], seq_h[b, vl-1])
__global__ void ctx_kernel(const int* uid, const int* vl, const float* user_emb,
                           const float* seqh, float* ctx) {
    const int b = blockIdx.x, dd = threadIdx.x;   // 384 threads
    float v;
    if (dd < 128) v = user_emb[(long)(uid[b] - 1) * DD_ + dd];
    else          v = seqh[((long)b * LL_ + (vl[b] - 1)) * DM_ + (dd - 128)];
    ctx[(long)b * 384 + dd] = v;
}

// entropy features from last-layer softmax attention (head-mean)
__global__ void entropy_kernel(const float* attn, const int* vl, float* ef) {
    const int b = blockIdx.x;
    const int lane = threadIdx.x;                 // one wave
    const int v = vl[b];
    float ents[2];
    for (int t = 0; t < 2; ++t) {
        const int pos = v - 1 - t;
        float acc = 0.0f;
        for (int j = lane; j < LL_; j += 32) {
            const float a0 = attn[(((long)b * 2 + 0) * LL_ + pos) * LL_ + j];
            const float a1 = attn[(((long)b * 2 + 1) * LL_ + pos) * LL_ + j];
            const float a = 0.5f * (a0 + a1) + 1e-10f;
            acc -= a * __logf(a);
        }
        for (int o = 16; o > 0; o >>= 1) acc += __shfl_down(acc, o);
        ents[t] = __shfl(acc, 0);
    }
    if (lane == 0) {
        const float maxH = fmaxf(__logf((float)v), 1.0f);
        ef[b * 3 + 0] = ents[1] / maxH;
        ef[b * 3 + 1] = ents[1] / maxH;
        ef[b * 3 + 2] = ents[0] / maxH;
    }
}

// ---------------------------------------------------------------------------
// host helpers
// ---------------------------------------------------------------------------
static inline void launch_gemm(hipStream_t st,
    const float* A, long sam, long sak,
    const float* B, long sbk, long sbn,
    float* C, long scm, long scn,
    const float* bias, int M, int N, int K, float alpha, int flags,
    int by = 1, int bz = 1,
    long bAy = 0, long bAz = 0, long bBy = 0, long bBz = 0,
    long bCy = 0, long bCz = 0)
{
    GemmDesc d;
    d.A = A; d.B = B; d.C = C; d.bias = bias;
    d.sam = sam; d.sak = sak; d.sbk = sbk; d.sbn = sbn; d.scm = scm; d.scn = scn;
    d.bAy = bAy; d.bAz = bAz; d.bBy = bBy; d.bBz = bBz; d.bCy = bCy; d.bCz = bCz;
    d.M = M; d.N = N; d.K = K; d.flags = flags; d.alpha = alpha;
    const long strips = (long)((M + 15) / 16) * (((N + 15) / 16 + 3) / 4);
    long blocks = (strips + 7) / 8;
    if (blocks > 2048) blocks = 2048;
    if (blocks < 1) blocks = 1;
    dim3 grid((unsigned)blocks, by, bz);
    const bool ak1 = (sak == 1), bk1 = (sbk == 1);
    if (flags == GF_RELU) {
        if (ak1 && bk1)      gemm_wmma_kernel<GF_RELU, true,  true ><<<grid, 256, 0, st>>>(d);
        else if (ak1)        gemm_wmma_kernel<GF_RELU, true,  false><<<grid, 256, 0, st>>>(d);
        else if (bk1)        gemm_wmma_kernel<GF_RELU, false, true ><<<grid, 256, 0, st>>>(d);
        else                 gemm_wmma_kernel<GF_RELU, false, false><<<grid, 256, 0, st>>>(d);
    } else if (flags == (GF_ACCUM | GF_RELU_A)) {
        if (ak1 && bk1)      gemm_wmma_kernel<GF_ACCUM | GF_RELU_A, true,  true ><<<grid, 256, 0, st>>>(d);
        else if (ak1)        gemm_wmma_kernel<GF_ACCUM | GF_RELU_A, true,  false><<<grid, 256, 0, st>>>(d);
        else if (bk1)        gemm_wmma_kernel<GF_ACCUM | GF_RELU_A, false, true ><<<grid, 256, 0, st>>>(d);
        else                 gemm_wmma_kernel<GF_ACCUM | GF_RELU_A, false, false><<<grid, 256, 0, st>>>(d);
    } else {
        if (ak1 && bk1)      gemm_wmma_kernel<0, true,  true ><<<grid, 256, 0, st>>>(d);
        else if (ak1)        gemm_wmma_kernel<0, true,  false><<<grid, 256, 0, st>>>(d);
        else if (bk1)        gemm_wmma_kernel<0, false, true ><<<grid, 256, 0, st>>>(d);
        else                 gemm_wmma_kernel<0, false, false><<<grid, 256, 0, st>>>(d);
    }
}

static inline void launch_fill(hipStream_t st, float* p, long n) {
    long blocks = (n + 255) / 256; if (blocks > 4096) blocks = 4096;
    fill_kernel<<<(unsigned)blocks, 256, 0, st>>>(p, n, 0.0f);
}

static inline void launch_scatter(hipStream_t st, const int* src, const int* dst,
                                  const float* w, int E, const float* H,
                                  int rowOff, float* agg) {
    long total = (long)E * 32;
    long blocks = (total + 255) / 256; if (blocks > 8192) blocks = 8192;
    scatter_add_kernel<<<(unsigned)blocks, 256, 0, st>>>(src, dst, w, E, H, rowOff, agg);
}

extern "C" void kernel_launch(void* const* d_in, const int* in_sizes, int n_in,
                              void* d_out, int out_size, void* d_ws, size_t ws_size,
                              hipStream_t stream) {
    (void)in_sizes; (void)out_size; (void)ws_size;
    if (n_in < 62) return;

    // ---- params (flattened in setup_inputs insertion order) ----
    const float* loc_emb   = (const float*)d_in[0];   // (20001,128)
    const float* time_emb  = (const float*)d_in[1];   // (49,128)
    const float* user_emb  = (const float*)d_in[2];   // (5000,128)
    const float* user_nr   = (const float*)d_in[3];   // (5000,128)
    const float* proto_emb = (const float*)d_in[4];   // (64,128)
    const float* W_user_q  = (const float*)d_in[5];
    const float* W_proto_k = (const float*)d_in[6];
    const float* W_proto_q = (const float*)d_in[7];
    const float* ulW[2] = { (const float*)d_in[8],  (const float*)d_in[10] };
    const float* ulB[2] = { (const float*)d_in[9],  (const float*)d_in[11] };
    const float* plW = (const float*)d_in[12];
    const float* plB = (const float*)d_in[13];
    const int layBase[2] = { 14, 30 };                // 16 leaves per enc layer
    const float* enc_g = (const float*)d_in[46];
    const float* enc_b = (const float*)d_in[47];
    const float* cls_W1 = (const float*)d_in[48];     // (384,768)
    const float* cls_b1 = (const float*)d_in[49];
    const float* cls_W2 = (const float*)d_in[50];     // (768,20001)
    const float* cls_b2 = (const float*)d_in[51];
    const float* ll_w = (const float*)d_in[52];
    const float* ul_w = (const float*)d_in[53];
    const float* pe   = (const float*)d_in[54];       // (512,256)
    const int* uid      = (const int*)d_in[55];
    const int* loc_seq  = (const int*)d_in[56];
    const int* time_seq = (const int*)d_in[57];
    // d_in[58] = attention_mask (derived from valid_len instead)
    const int* valid_len = (const int*)d_in[59];
    const int* ll_src = (const int*)d_in[60];
    const int* ll_dst = ll_src + E_LL_;
    const int* ul_src = (const int*)d_in[61];
    const int* ul_dst = ul_src + E_UL_;

    float* ws = (float*)d_ws;
    float* out_logits  = (float*)d_out;                       // 64*20001
    float* out_plgnn   = out_logits + (long)BB_ * 20001;      // 20064*128
    float* out_entropy = out_plgnn + (long)N_PL_ * DD_;       // 64*3

    // ---- GNN-phase arena (aliased later by transformer arena) ----
    float* userq  = ws + 0;                 // 5000*128
    float* protok = userq + 640000;         // 64*128
    float* logits = protok + 8192;          // 5000*64
    float* puw    = logits + 320000;        // 5000 (+pad)
    int*   hidx   = (int*)(puw + 5120);     // 5000 ints (+pad)
    float* hbuf   = (float*)hidx + 5120;    // 25064*128
    float* agg    = hbuf + (long)N_BI_ * DD_;
    float* lk     = agg + (long)N_BI_ * DD_;   // 20000*128
    float* pqb    = lk + (long)NUM_LOC_ * DD_; // 64*128
    float* score  = pqb + 8192;                // 64*20000
    float* agg2   = score + (long)NUM_PROTO_ * NUM_LOC_;  // 20064*128

    // ================= phase 1: proto attention =================
    launch_gemm(stream, user_nr, DD_, 1, W_user_q, DD_, 1, userq, DD_, 1,
                nullptr, NUM_USER_, DD_, DD_, 1.0f, 0);
    launch_gemm(stream, proto_emb, DD_, 1, W_proto_k, DD_, 1, protok, DD_, 1,
                nullptr, NUM_PROTO_, DD_, DD_, 1.0f, 0);
    // logits = user_q @ proto_k^T / sqrt(D)
    launch_gemm(stream, userq, DD_, 1, protok, 1, DD_, logits, 64, 1,
                nullptr, NUM_USER_, NUM_PROTO_, DD_, 0.088388347648318447f, 0);
    proto_attn_kernel<<<(NUM_USER_ + 7) / 8, 256, 0, stream>>>(logits, puw, hidx);

    // ================= phase 2: bipartite GNN (2 layers) =================
    build_h_kernel<<<4096, 256, 0, stream>>>(loc_emb, user_nr, proto_emb, hbuf);
    for (int l = 0; l < 2; ++l) {
        launch_fill(stream, agg, (long)N_BI_ * DD_);
        launch_scatter(stream, ul_src, ul_dst, ul_w, E_UL_, hbuf, 0, agg);
        launch_scatter(stream, ll_src, ll_dst, ll_w, E_LL_, hbuf, 0, agg);
        scatter_pu_kernel<<<NUM_USER_, DD_, 0, stream>>>(hidx, puw, hbuf, agg);
        launch_gemm(stream, agg, DD_, 1, ulW[l], DD_, 1, hbuf, DD_, 1,
                    ulB[l], N_BI_, DD_, DD_, 1.0f, GF_RELU);
    }
    // proto_q = bi_proto @ W_proto_q_sem ; normalize rows; lk = normalize(bi_loc)
    launch_gemm(stream, hbuf + (long)(NUM_LOC_ + NUM_USER_) * DD_, DD_, 1,
                W_proto_q, DD_, 1, pqb, DD_, 1, nullptr, NUM_PROTO_, DD_, DD_, 1.0f, 0);
    rownorm_kernel<<<(NUM_PROTO_ + 7) / 8, 256, 0, stream>>>(pqb, pqb, NUM_PROTO_);
    rownorm_kernel<<<(NUM_LOC_ + 7) / 8, 256, 0, stream>>>(hbuf, lk, NUM_LOC_);
    // score = pq @ lk^T   (64 x 20000)
    launch_gemm(stream, pqb, DD_, 1, lk, 1, DD_, score, NUM_LOC_, 1,
                nullptr, NUM_PROTO_, NUM_LOC_, DD_, 1.0f, 0);

    // ================= phase 3: pl-GNN =================
    launch_fill(stream, agg2, (long)N_PL_ * DD_);
    launch_scatter(stream, ll_src, ll_dst, ll_w, E_LL_, loc_emb, 1, agg2);
    // agg2[:NUM_LOC] += relu(score)^T @ proto_emb   (TN, relu on A)
    launch_gemm(stream, score, 1, NUM_LOC_, proto_emb, DD_, 1, agg2, DD_, 1,
                nullptr, NUM_LOC_, DD_, NUM_PROTO_, 1.0f, GF_ACCUM | GF_RELU_A);
    // plgnn_output = relu(agg2 @ W + b) -> d_out slot 2
    launch_gemm(stream, agg2, DD_, 1, plW, DD_, 1, out_plgnn, DD_, 1,
                plB, N_PL_, DD_, DD_, 1.0f, GF_RELU);

    // ================= phase 4: transformer (arena aliases GNN arena) ====
    const long TSZ = (long)BB_ * LL_ * DM_;        // 2097152
    float* x  = ws + 0;
    float* qb = x + TSZ;
    float* kb = qb + TSZ;
    float* vb = kb + TSZ;
    float* sb = vb + TSZ;                          // B*NH*L*L
    float* ob = sb + TSZ;
    float* t1 = ob + TSZ;
    float* ctx  = t1 + TSZ;                        // 64*384
    float* hcls = ctx + 24576;                     // 64*768

    embed_kernel<<<(unsigned)((TSZ + 255) / 256), 256, 0, stream>>>(
        loc_seq, time_seq, loc_emb, time_emb, pe, x);

    const int rows = BB_ * LL_;                    // 8192
    for (int l = 0; l < 2; ++l) {
        const int b0 = layBase[l];
        const float* Wq = (const float*)d_in[b0 + 0]; const float* bq = (const float*)d_in[b0 + 1];
        const float* Wk = (const float*)d_in[b0 + 2]; const float* bk = (const float*)d_in[b0 + 3];
        const float* Wv = (const float*)d_in[b0 + 4]; const float* bv = (const float*)d_in[b0 + 5];
        const float* Wo = (const float*)d_in[b0 + 6]; const float* bo = (const float*)d_in[b0 + 7];
        const float* W1 = (const float*)d_in[b0 + 8]; const float* b1 = (const float*)d_in[b0 + 9];
        const float* W2 = (const float*)d_in[b0 + 10]; const float* b2 = (const float*)d_in[b0 + 11];
        const float* g1 = (const float*)d_in[b0 + 12]; const float* be1 = (const float*)d_in[b0 + 13];
        const float* g2 = (const float*)d_in[b0 + 14]; const float* be2 = (const float*)d_in[b0 + 15];

        launch_gemm(stream, x, DM_, 1, Wq, DM_, 1, qb, DM_, 1, bq, rows, DM_, DM_, 1.0f, 0);
        launch_gemm(stream, x, DM_, 1, Wk, DM_, 1, kb, DM_, 1, bk, rows, DM_, DM_, 1.0f, 0);
        launch_gemm(stream, x, DM_, 1, Wv, DM_, 1, vb, DM_, 1, bv, rows, DM_, DM_, 1.0f, 0);
        // s[b,h] = Q K^T / sqrt(DH) : batched over (y=b, z=h)
        launch_gemm(stream, qb, DM_, 1, kb, 1, DM_, sb, LL_, 1, nullptr,
                    LL_, LL_, DH_, 0.088388347648318447f, 0, BB_, NH_,
                    (long)LL_ * DM_, DH_, (long)LL_ * DM_, DH_,
                    (long)NH_ * LL_ * LL_, (long)LL_ * LL_);
        attn_softmax_kernel<<<(BB_ * NH_ * LL_) / 8, 256, 0, stream>>>(sb, valid_len);
        // o[b,h] = A V : batched
        launch_gemm(stream, sb, LL_, 1, vb, DM_, 1, ob, DM_, 1, nullptr,
                    LL_, DH_, LL_, 1.0f, 0, BB_, NH_,
                    (long)NH_ * LL_ * LL_, (long)LL_ * LL_,
                    (long)LL_ * DM_, DH_, (long)LL_ * DM_, DH_);
        launch_gemm(stream, ob, DM_, 1, Wo, DM_, 1, t1, DM_, 1, bo, rows, DM_, DM_, 1.0f, 0);
        layernorm_kernel<<<rows / 8, 256, 0, stream>>>(x, t1, x, g1, be1, rows);
        launch_gemm(stream, x, DM_, 1, W1, DM_, 1, qb, DM_, 1, b1, rows, DM_, DM_, 1.0f, GF_RELU);
        launch_gemm(stream, qb, DM_, 1, W2, DM_, 1, kb, DM_, 1, b2, rows, DM_, DM_, 1.0f, 0);
        layernorm_kernel<<<rows / 8, 256, 0, stream>>>(x, kb, x, g2, be2, rows);
    }
    // seq_h = LN(x) ; sb still holds last layer's softmaxed attention
    layernorm_kernel<<<rows / 8, 256, 0, stream>>>(x, nullptr, t1, enc_g, enc_b, rows);

    // ================= phase 5: head =================
    ctx_kernel<<<BB_, 384, 0, stream>>>(uid, valid_len, user_emb, t1, ctx);
    launch_gemm(stream, ctx, 384, 1, cls_W1, 768, 1, hcls, 768, 1,
                cls_b1, BB_, 768, 384, 1.0f, GF_RELU);
    launch_gemm(stream, hcls, 768, 1, cls_W2, 20001, 1, out_logits, 20001, 1,
                cls_b2, BB_, 20001, 768, 1.0f, 0);
    entropy_kernel<<<BB_, 32, 0, stream>>>(sb, valid_len, out_entropy);
}
